// Block_31542239822107
// MI455X (gfx1250) — matrix operand order repaired and checked by reading, not compiled, not run on hardware
//
#include <hip/hip_runtime.h>
#include <cstdint>
#include <cstddef>

// ---------------- problem constants ----------------
constexpr int Bb  = 32;
constexpr int Ss  = 577;
constexpr int Dd  = 768;
constexpr int Hh  = 12;
constexpr int DHh = 64;
constexpr int Ff  = 3072;
constexpr int SP  = 640;               // 577 rounded up to 10*64: all K-loops stage in BK=64
constexpr long BS  = (long)Bb * Ss;    // 18464 rows
constexpr long BSP = 18496;            // BS rounded up to 64 (tile M)

typedef __bf16 bf16;
typedef __attribute__((ext_vector_type(8)))  __bf16 bf16x8;
typedef __attribute__((ext_vector_type(16))) __bf16 bf16x16;
typedef __attribute__((ext_vector_type(8)))  float  floatx8;

// ---------------- CDNA5 async global->LDS copy (ASYNCcnt tracked) ----------
__device__ __forceinline__ void async_copy_b128(uint32_t lds_off, const void* gptr) {
  asm volatile("global_load_async_to_lds_b128 %0, %1, off"
               :: "v"(lds_off), "v"((uint64_t)(uintptr_t)gptr)
               : "memory");
}
template <int N>
__device__ __forceinline__ void wait_async_le() {
  asm volatile("s_wait_asynccnt %0" :: "n"(N) : "memory");
}
__device__ __forceinline__ uint32_t lds_off32(const void* p) {
  return (uint32_t)(uintptr_t)p;   // low 32 bits of shared-aperture flat addr = LDS byte offset
}

// ---------------- weight conversion kernels ----------------
// generic transpose+cast: in row-major [K][N] f32 -> out row-major [N][K] bf16
__global__ void transpose_to_bf16_kernel(const float* __restrict__ in,
                                         bf16* __restrict__ out, int K, int N) {
  int i = blockIdx.x * blockDim.x + threadIdx.x;
  if (i < K * N) {
    int n = i / K, k = i % K;
    out[i] = (bf16)in[(size_t)k * N + n];
  }
}
// Wq/Wk/Wv [H, D, DH] -> B^T layout [N=H*DH][K=D]
__global__ void permute_qkvw_kernel(const float* __restrict__ in,
                                    bf16* __restrict__ out) {
  int i = blockIdx.x * blockDim.x + threadIdx.x;
  if (i < Dd * Dd) {
    int c = i / Dd;               // output row  n = h*64 + kk
    int d = i % Dd;               // output col  k = d
    out[i] = (bf16)in[(size_t)(c >> 6) * (Dd * DHh) + (size_t)d * DHh + (c & 63)];
  }
}

// ---------------- LayerNorm (f32 in -> bf16 out) ----------------
__global__ void ln_kernel(const float* __restrict__ x,
                          const float* __restrict__ g,
                          const float* __restrict__ b,
                          bf16* __restrict__ out) {
  const int row = blockIdx.x;
  const float* xr = x + (size_t)row * Dd;
  float s = 0.f, s2 = 0.f;
  for (int i = threadIdx.x; i < Dd; i += 256) {
    float v = xr[i];
    s += v; s2 += v * v;
  }
  __shared__ float sh[256];
  __shared__ float sh2[256];
  sh[threadIdx.x] = s; sh2[threadIdx.x] = s2;
  __syncthreads();
  for (int off = 128; off > 0; off >>= 1) {
    if (threadIdx.x < off) {
      sh[threadIdx.x]  += sh[threadIdx.x + off];
      sh2[threadIdx.x] += sh2[threadIdx.x + off];
    }
    __syncthreads();
  }
  float mu  = sh[0] * (1.0f / Dd);
  float var = sh2[0] * (1.0f / Dd) - mu * mu;
  float rs  = rsqrtf(var + 1e-5f);
  for (int i = threadIdx.x; i < Dd; i += 256)
    out[(size_t)row * Dd + i] = (bf16)((xr[i] - mu) * rs * g[i] + b[i]);
}

// ------- column softmax over QUERY axis (faithful quirk), pads cols with 0 ----
__global__ void colsoftmax_kernel(const float* __restrict__ scores,
                                  bf16* __restrict__ probs) {
  int t = blockIdx.x * blockDim.x + threadIdx.x;
  int h = blockIdx.y;
  if (t >= SP) return;
  bf16* pr = probs + (size_t)h * Ss * SP + t;
  if (t >= Ss) {                       // zero pad columns so AV K-padding is exact
    for (int s = 0; s < Ss; ++s) pr[(size_t)s * SP] = (bf16)0.0f;
    return;
  }
  const float* sc = scores + (size_t)h * Ss * SP + t;
  float m = -3.0e38f, sum = 0.f;
  for (int s = 0; s < Ss; ++s) {
    float v = sc[(size_t)s * SP];
    if (v > m) { sum = sum * __expf(m - v) + 1.0f; m = v; }
    else       { sum += __expf(v - m); }
  }
  float inv = 1.0f / sum;
  for (int s = 0; s < Ss; ++s) {
    float v = sc[(size_t)s * SP];
    pr[(size_t)s * SP] = (bf16)(__expf(v - m) * inv);
  }
}

// ---------------- bf16 WMMA GEMM:  C[z] = act((A[z] * Bt[z]^T + bias)*scale) (+resid)
// A  : row-major [M][K] (rows padded to tile — no load guards)
// Bt : row-major [N][K] (B transposed; rows padded — no load guards)
// Tile: 64(M) x 128(N) x 64(K-stage); 8 wave32s = 4 Mstrips x 2 Ncols;
// 8 v_wmma_f32_16x16x32_bf16 per wave per stage.
// Double-buffered staging via global_load_async_to_lds_b128 (ASYNCcnt):
// stage i+1 DMA overlaps stage i WMMA; s_wait_asynccnt 6 drains only the old group.
// K must be a multiple of 64.
// OMODE: 0 f32 row-major | 1 bf16 row-major | 2 bf16 scatter [B,H,S,DH] | 3 bf16 scatter V^T
template <bool BIAS, bool RESID, bool GELUF, int OMODE>
__global__ void gemm_kernel(const bf16* __restrict__ A, int lda, long long sAz,
                            const bf16* __restrict__ Bt, int ldb, long long sBz,
                            const float* __restrict__ bias,
                            const float* __restrict__ resid, int ldr,
                            void* __restrict__ out, int ldo, long long sCz,
                            int M, int N, int K, float scale) {
  __shared__ __align__(16) bf16 sA[2][64 * 64];    // [m][k]
  __shared__ __align__(16) bf16 sB[2][128 * 64];   // [n][k]

  const int z = blockIdx.z;
  A  += (size_t)z * sAz;
  Bt += (size_t)z * sBz;

  const int m0   = blockIdx.y * 64;
  const int n0   = blockIdx.x * 128;
  const int tid  = threadIdx.x;
  const int lane = tid & 31;
  const int wave = tid >> 5;
  const int l16  = lane & 15;
  const int lhi  = lane >> 4;
  const int wm   = wave & 3;    // M strip (16 rows)
  const int wn   = wave >> 2;   // N half (64 cols)

  // staging assignments: 16 k-elements (two b128 copies) per (row, chunk)
  const int am  = tid >> 2;
  const int bn1 = tid >> 2;
  const int bn2 = (tid + 256) >> 2;
  const int kb  = (tid & 3) * 16;
  const bf16* agp  = A  + (size_t)(m0 + am)  * lda + kb;
  const bf16* bgp1 = Bt + (size_t)(n0 + bn1) * ldb + kb;
  const bf16* bgp2 = Bt + (size_t)(n0 + bn2) * ldb + kb;
  const uint32_t aoff  = lds_off32(&sA[0][am  * 64 + kb]);
  const uint32_t boff1 = lds_off32(&sB[0][bn1 * 64 + kb]);
  const uint32_t boff2 = lds_off32(&sB[0][bn2 * 64 + kb]);
  constexpr uint32_t A_BUF = 64 * 64 * 2;     // bytes per A buffer
  constexpr uint32_t B_BUF = 128 * 64 * 2;    // bytes per B buffer

  auto issue_stage = [&](int buf, int k0) {
    const uint32_t ab = aoff  + (uint32_t)buf * A_BUF;
    const uint32_t b1 = boff1 + (uint32_t)buf * B_BUF;
    const uint32_t b2 = boff2 + (uint32_t)buf * B_BUF;
    async_copy_b128(ab,      agp  + k0);
    async_copy_b128(ab + 16, agp  + k0 + 8);
    async_copy_b128(b1,      bgp1 + k0);
    async_copy_b128(b1 + 16, bgp1 + k0 + 8);
    async_copy_b128(b2,      bgp2 + k0);
    async_copy_b128(b2 + 16, bgp2 + k0 + 8);
  };

  floatx8 acc[4] = {};
  const int nstages = K >> 6;

  issue_stage(0, 0);
  for (int s = 0; s < nstages; ++s) {
    const int cur = s & 1;
    const bool hasNext = (s + 1 < nstages);
    if (hasNext) {
      issue_stage(1 - cur, (s + 1) << 6);
      wait_async_le<6>();     // drain previous stage's copies; keep new 6 in flight
    } else {
      wait_async_le<0>();
    }
    __syncthreads();          // all waves' copies for `cur` visible

    const bf16* As = &sA[cur][0];
    const bf16* Bs = &sB[cur][0];
    const int mrow = wm * 16 + l16;
#pragma unroll
    for (int kk = 0; kk < 64; kk += 32) {
      bf16x8 alo = *(const bf16x8*)&As[mrow * 64 + kk + lhi * 8];
      bf16x8 ahi = *(const bf16x8*)&As[mrow * 64 + kk + 16 + lhi * 8];
      bf16x16 afrag = __builtin_shufflevector(alo, ahi,
          0, 1, 2, 3, 4, 5, 6, 7, 8, 9, 10, 11, 12, 13, 14, 15);
#pragma unroll
      for (int sub = 0; sub < 4; ++sub) {
        const int nrow = wn * 64 + sub * 16 + l16;
        bf16x8 blo = *(const bf16x8*)&Bs[nrow * 64 + kk + lhi * 8];
        bf16x8 bhi = *(const bf16x8*)&Bs[nrow * 64 + kk + 16 + lhi * 8];
        bf16x16 bfrag = __builtin_shufflevector(blo, bhi,
            0, 1, 2, 3, 4, 5, 6, 7, 8, 9, 10, 11, 12, 13, 14, 15);
        acc[sub] = __builtin_amdgcn_wmma_f32_16x16x32_bf16(
            false, afrag, false, bfrag, (short)0, acc[sub], false, false);
      }
    }
    __syncthreads();          // reads of `cur` done before it is re-filled
  }

  // epilogue (C f32 layout: VGPR r -> M = r + lhi*8)
#pragma unroll
  for (int r = 0; r < 8; ++r) {
    const int mg = m0 + wm * 16 + lhi * 8 + r;
#pragma unroll
    for (int sub = 0; sub < 4; ++sub) {
      const int ng = n0 + wn * 64 + sub * 16 + l16;
      float v = acc[sub][r];
      if (mg < M && ng < N) {
        if (BIAS)  v += bias[ng];
        v *= scale;
        if (GELUF) v = 0.5f * v * (1.0f + erff(v * 0.70710678118654752f));
        if (RESID) v += resid[(size_t)mg * ldr + ng];
        if (OMODE == 0) {
          ((float*)out)[(size_t)z * sCz + (size_t)mg * ldo + ng] = v;
        } else if (OMODE == 1) {
          ((bf16*)out)[(size_t)z * sCz + (size_t)mg * ldo + ng] = (bf16)v;
        } else if (OMODE == 2) {   // q/k scatter to [B,H,S,DH]
          int bidx = mg / Ss, sidx = mg % Ss;
          int hh = ng >> 6, dh = ng & 63;
          ((bf16*)out)[(((size_t)bidx * Hh + hh) * Ss + sidx) * DHh + dh] = (bf16)v;
        } else {                   // v scatter to V^T  [B,H,DH][SP]
          int bidx = mg / Ss, sidx = mg % Ss;
          int hh = ng >> 6, dh = ng & 63;
          ((bf16*)out)[(((size_t)bidx * Hh + hh) * DHh + dh) * SP + sidx] = (bf16)v;
        }
      }
    }
  }
}

// ---------------- host launcher ----------------
extern "C" void kernel_launch(void* const* d_in, const int* in_sizes, int n_in,
                              void* d_out, int out_size, void* d_ws, size_t ws_size,
                              hipStream_t stream) {
  (void)in_sizes; (void)n_in; (void)out_size; (void)ws_size;

  const float* x    = (const float*)d_in[0];
  const float* Wq   = (const float*)d_in[1];
  const float* bq   = (const float*)d_in[2];
  const float* Wk   = (const float*)d_in[3];
  const float* bk   = (const float*)d_in[4];
  const float* Wv   = (const float*)d_in[5];
  const float* bv   = (const float*)d_in[6];
  const float* Wo   = (const float*)d_in[7];
  const float* bo   = (const float*)d_in[8];
  const float* ln1g = (const float*)d_in[9];
  const float* ln1b = (const float*)d_in[10];
  const float* ln2g = (const float*)d_in[11];
  const float* ln2b = (const float*)d_in[12];
  const float* W1   = (const float*)d_in[13];
  const float* b1   = (const float*)d_in[14];
  const float* W2   = (const float*)d_in[15];
  const float* b2   = (const float*)d_in[16];
  float* out = (float*)d_out;

  // ---- carve workspace (all operands padded so GEMM loads need no guards) ----
  uint8_t* wp = (uint8_t*)d_ws;
  auto carve = [&](size_t bytes) -> void* {
    void* p = (void*)wp;
    wp += (bytes + 255) & ~(size_t)255;
    return p;
  };
  const size_t headRows = (size_t)Bb * Hh * Ss;      // rows of [B,H,S,·]
  bf16* Wqt = (bf16*)carve((size_t)Dd * Dd * 2);
  bf16* Wkt = (bf16*)carve((size_t)Dd * Dd * 2);
  bf16* Wvt = (bf16*)carve((size_t)Dd * Dd * 2);
  bf16* Wot = (bf16*)carve((size_t)Dd * Dd * 2);
  bf16* W1t = (bf16*)carve((size_t)Dd * Ff * 2);     // [F][D]
  bf16* W2t = (bf16*)carve((size_t)Ff * Dd * 2);     // [D][F]
  bf16* hb  = (bf16*)carve((size_t)BSP * Dd * 2);
  bf16* qb  = (bf16*)carve((headRows + 64) * DHh * 2);
  bf16* kbw = (bf16*)carve((headRows + 64) * DHh * 2);
  size_t vT_bytes = ((size_t)Bb * Hh * DHh + 64) * SP * 2;
  bf16* vT  = (bf16*)carve(vT_bytes);                // [B,H,DH][SP]
  bf16* cat = (bf16*)carve((size_t)BSP * Dd * 2);
  float* x1 = (float*)carve((size_t)BS * Dd * 4);
  bf16* h2b = (bf16*)carve((size_t)BSP * Dd * 2);
  bf16* a1b = (bf16*)carve((size_t)BSP * Ff * 2);
  float* sc = (float*)carve((size_t)Hh * Ss * SP * 4);
  bf16* pr  = (bf16*)carve(((size_t)Hh * Ss + 64) * SP * 2);

  const dim3 blk(256);

  // vT pad columns must be real zeros (multiplied by zero probs pad cols)
  hipMemsetAsync(vT, 0, vT_bytes, stream);

  // ---- weight conversion (all B operands stored transposed [N][K]) ----
  {
    int n = Dd * Dd;
    dim3 g((n + 255) / 256);
    permute_qkvw_kernel<<<g, blk, 0, stream>>>(Wq, Wqt);
    permute_qkvw_kernel<<<g, blk, 0, stream>>>(Wk, Wkt);
    permute_qkvw_kernel<<<g, blk, 0, stream>>>(Wv, Wvt);
    transpose_to_bf16_kernel<<<g, blk, 0, stream>>>(Wo, Wot, Dd, Dd);
    int nf = Dd * Ff;
    dim3 gf((nf + 255) / 256);
    transpose_to_bf16_kernel<<<gf, blk, 0, stream>>>(W1, W1t, Dd, Ff);  // -> [F][D]
    transpose_to_bf16_kernel<<<gf, blk, 0, stream>>>(W2, W2t, Ff, Dd);  // -> [D][F]
  }

  // ---- LN1 ----
  ln_kernel<<<dim3((unsigned)BS), blk, 0, stream>>>(x, ln1g, ln1b, hb);

  // ---- QKV projections (q pre-scaled by 1/sqrt(DH)) ----
  {
    dim3 g(Dd / 128, (unsigned)(BSP / 64), 1);
    gemm_kernel<true, false, false, 2><<<g, blk, 0, stream>>>(
        hb, Dd, 0, Wqt, Dd, 0, bq, nullptr, 0, qb, 0, 0,
        (int)BS, Dd, Dd, 0.125f);
    gemm_kernel<true, false, false, 2><<<g, blk, 0, stream>>>(
        hb, Dd, 0, Wkt, Dd, 0, bk, nullptr, 0, kbw, 0, 0,
        (int)BS, Dd, Dd, 1.0f);
    gemm_kernel<true, false, false, 3><<<g, blk, 0, stream>>>(
        hb, Dd, 0, Wvt, Dd, 0, bv, nullptr, 0, vT, 0, 0,
        (int)BS, Dd, Dd, 1.0f);
  }

  // ---- attention, chunked per batch (scores scratch reused) ----
  for (int b = 0; b < Bb; ++b) {
    const bf16* qB = qb  + (size_t)b * Hh * Ss * DHh;
    const bf16* kB = kbw + (size_t)b * Hh * Ss * DHh;
    const bf16* vB = vT  + (size_t)b * Hh * DHh * SP;
    bf16* catB = cat + (size_t)b * Ss * Dd;

    // scores[h] = q[h] (S x 64) * k[h]^T -> f32 [S x S] (row stride SP)
    {
      dim3 g((Ss + 127) / 128, (Ss + 63) / 64, Hh);
      gemm_kernel<false, false, false, 0><<<g, blk, 0, stream>>>(
          qB, DHh, (long long)Ss * DHh,
          kB, DHh, (long long)Ss * DHh,
          nullptr, nullptr, 0,
          sc, SP, (long long)Ss * SP,
          Ss, Ss, DHh, 1.0f);
    }
    // softmax over query axis (columns); zero-fills pad columns [Ss,SP)
    colsoftmax_kernel<<<dim3((SP + 255) / 256, Hh), blk, 0, stream>>>(sc, pr);
    // attn[h] = probs[h] (S x SP) * vT[h]^T (SP x 64) -> bf16 into concat layout
    {
      dim3 g(1, (Ss + 63) / 64, Hh);
      gemm_kernel<false, false, false, 1><<<g, blk, 0, stream>>>(
          pr, SP, (long long)Ss * SP,
          vB, SP, (long long)DHh * SP,
          nullptr, nullptr, 0,
          catB, Dd, (long long)DHh,   // z offset = h*64 columns
          Ss, DHh, SP, 1.0f);
    }
  }

  // ---- Wo projection + residual: x1 = x + concat @ Wo + bo ----
  {
    dim3 g(Dd / 128, (unsigned)(BSP / 64), 1);
    gemm_kernel<true, true, false, 0><<<g, blk, 0, stream>>>(
        cat, Dd, 0, Wot, Dd, 0, bo, x, Dd, x1, Dd, 0,
        (int)BS, Dd, Dd, 1.0f);
  }

  // ---- LN2 ----
  ln_kernel<<<dim3((unsigned)BS), blk, 0, stream>>>(x1, ln2g, ln2b, h2b);

  // ---- MLP1: gelu(h2 @ W1 + b1) -> bf16 ----
  {
    dim3 g(Ff / 128, (unsigned)(BSP / 64), 1);
    gemm_kernel<true, false, true, 1><<<g, blk, 0, stream>>>(
        h2b, Dd, 0, W1t, Dd, 0, b1, nullptr, 0, a1b, Ff, 0,
        (int)BS, Ff, Dd, 1.0f);
  }

  // ---- MLP2 + residual: out = x1 + a1 @ W2 + b2 ----
  {
    dim3 g(Dd / 128, (unsigned)(BSP / 64), 1);
    gemm_kernel<true, true, false, 0><<<g, blk, 0, stream>>>(
        a1b, Ff, 0, W2t, Ff, 0, b2, x1, Dd, out, Dd, 0,
        (int)BS, Dd, Ff, 1.0f);
  }
}
